// Validator_76227079570085
// MI455X (gfx1250) — matrix operand order repaired and checked by reading, not compiled, not run on hardware
//
#include <hip/hip_runtime.h>
#include <hip/hip_bf16.h>

// ---------------- types for WMMA fragments (gfx1250, wave32) ----------------
typedef _Float16 v16h __attribute__((ext_vector_type(16)));
typedef _Float16 h8   __attribute__((ext_vector_type(8)));
typedef float    v8f  __attribute__((ext_vector_type(8)));
typedef int      gv4i __attribute__((vector_size(16)));   // matches builtin param V4i

#define S_DIM 127
#define S3    (127 * 127 * 127)
#define NPTS  200000

// ---------------- async global->LDS path (guarded; fallback = ds store) ----
#if defined(__has_builtin)
#  if __has_builtin(__builtin_amdgcn_global_load_async_to_lds_b128) && \
      __has_builtin(__builtin_amdgcn_s_wait_asynccnt)
#    define USE_ASYNC_LDS 1
#  endif
#endif
#ifndef USE_ASYNC_LDS
#  define USE_ASYNC_LDS 0
#endif

__device__ __forceinline__ void async_copy16(const _Float16* g, _Float16* s) {
#if USE_ASYNC_LDS
  __builtin_amdgcn_global_load_async_to_lds_b128(
      (__attribute__((address_space(1))) gv4i*)(g),
      (__attribute__((address_space(3))) gv4i*)(s), 0, 0);
#else
  *(h8*)s = *(const h8*)g;
#endif
}
__device__ __forceinline__ void async_wait_all() {
#if USE_ASYNC_LDS
  __builtin_amdgcn_s_wait_asynccnt(0);
#endif
}

// =====================================================================
// zero-fill (graph-capture-safe replacement for memset)
// =====================================================================
__global__ void k_zero(float* __restrict__ p, int n) {
  int i = blockIdx.x * blockDim.x + threadIdx.x;
  if (i < n) p[i] = 0.0f;
}

// =====================================================================
// voxelize: scatter point colors into grid + counts (atomic)
// =====================================================================
__global__ void k_scatter(const int* __restrict__ coords, const float* __restrict__ colors,
                          float* __restrict__ gridF, float* __restrict__ cnt, int npts) {
  int i = blockIdx.x * blockDim.x + threadIdx.x;
  if (i >= npts) return;
  int x = coords[i * 4 + 0], y = coords[i * 4 + 1], z = coords[i * 4 + 2], b = coords[i * 4 + 3];
  int lin = ((b * S_DIM + x) * S_DIM + y) * S_DIM + z;
  atomicAdd(&cnt[lin], 1.0f);
  atomicAdd(&gridF[lin * 3 + 0], colors[i * 3 + 0]);
  atomicAdd(&gridF[lin * 3 + 1], colors[i * 3 + 1]);
  atomicAdd(&gridF[lin * 3 + 2], colors[i * 3 + 2]);
}

__global__ void k_finalize(float* __restrict__ gridF, const float* __restrict__ cnt,
                           unsigned char* __restrict__ mask, int nv) {
  int i = blockIdx.x * blockDim.x + threadIdx.x;
  if (i >= nv) return;
  float c = cnt[i];
  if (c > 0.0f) {
    float inv = 1.0f / c;
    gridF[i * 3 + 0] *= inv;
    gridF[i * 3 + 1] *= inv;
    gridF[i * 3 + 2] *= inv;
    mask[i] = 1;
  } else {
    mask[i] = 0;  // gridF already zero
  }
}

// =====================================================================
// fused conv0 (3->32 subm, SAME) + sparse maxpool 127 -> 63
// grid is zero at inactive sites, so subm conv = plain conv at active sites.
// =====================================================================
__global__ __launch_bounds__(256)
void k_conv0_pool(const float* __restrict__ gridF, const unsigned char* __restrict__ m127,
                  const float* __restrict__ w0,  // [27][3][32]
                  float* __restrict__ x0, unsigned char* __restrict__ m63) {
  const int D = 63;
  const int D3 = D * D * D;
  int tid = blockIdx.x * blockDim.x + threadIdx.x;
  if (tid >= 2 * D3 * 32) return;
  int c = tid & 31;
  int pv = tid >> 5;
  int b = pv / D3;
  int r = pv - b * D3;
  int z = r / (D * D); r -= z * D * D;
  int y = r / D;
  int x = r - y * D;

  float best = -3.4e38f;
  bool any = false;
  for (int p = 0; p < 27; ++p) {
    int pz = z * 2 + p / 9, py = y * 2 + (p / 3) % 3, px = x * 2 + p % 3;
    int site = ((b * S_DIM + pz) * S_DIM + py) * S_DIM + px;
    if (!m127[site]) continue;
    any = true;
    float s = 0.0f;
    for (int q = 0; q < 27; ++q) {
      int qz = pz + q / 9 - 1, qy = py + (q / 3) % 3 - 1, qx = px + q % 3 - 1;
      if ((unsigned)qz >= (unsigned)S_DIM || (unsigned)qy >= (unsigned)S_DIM ||
          (unsigned)qx >= (unsigned)S_DIM) continue;
      int gi = (((b * S_DIM + qz) * S_DIM + qy) * S_DIM + qx) * 3;
      s = fmaf(gridF[gi + 0], w0[(q * 3 + 0) * 32 + c], s);
      s = fmaf(gridF[gi + 1], w0[(q * 3 + 1) * 32 + c], s);
      s = fmaf(gridF[gi + 2], w0[(q * 3 + 2) * 32 + c], s);
    }
    best = best > s ? best : s;
  }
  x0[pv * 32 + c] = any ? best : 0.0f;
  if (c == 0) m63[pv] = any ? 1 : 0;
}

// =====================================================================
// pool_mask: 3^3 window, stride 2, VALID (Din = 2*Dout+1)
// =====================================================================
__global__ void k_pool_mask(const unsigned char* __restrict__ mi, int Din,
                            unsigned char* __restrict__ mo, int Dout) {
  int tid = blockIdx.x * blockDim.x + threadIdx.x;
  int d3 = Dout * Dout * Dout;
  if (tid >= 2 * d3) return;
  int b = tid / d3;
  int r = tid - b * d3;
  int z = r / (Dout * Dout); r -= z * Dout * Dout;
  int y = r / Dout;
  int x = r - y * Dout;
  int ind3 = Din * Din * Din;
  unsigned char any = 0;
  for (int d = 0; d < 27; ++d) {
    int iz = z * 2 + d / 9, iy = y * 2 + (d / 3) % 3, ix = x * 2 + d % 3;
    any |= mi[b * ind3 + (iz * Din + iy) * Din + ix];
  }
  mo[tid] = any ? 1 : 0;
}

// =====================================================================
// BN statistics over active sites: per-block LDS partial reduction
// (ds_add_f32) then one global atomic per channel.
// stats layout: [0..C) sum, [C..2C) sumsq, [2C] active-count
// =====================================================================
__global__ __launch_bounds__(256)
void k_bn_stats(const float* __restrict__ xin, const unsigned char* __restrict__ mask,
                int Nvox, int C, float* __restrict__ stats) {
  __shared__ float sd[352];
  for (int i = threadIdx.x; i < 2 * C + 1; i += blockDim.x) sd[i] = 0.0f;
  __syncthreads();
  const int VPB = 64;
  int v0 = blockIdx.x * VPB;
  for (int idx = threadIdx.x; idx < VPB * C; idx += blockDim.x) {
    int v = v0 + idx / C;
    int c = idx % C;
    if (v < Nvox && mask[v]) {
      float t = xin[v * C + c];
      atomicAdd(&sd[c], t);
      atomicAdd(&sd[C + c], t * t);
    }
  }
  for (int idx = threadIdx.x; idx < VPB; idx += blockDim.x) {
    int v = v0 + idx;
    if (v < Nvox && mask[v]) atomicAdd(&sd[2 * C], 1.0f);
  }
  __syncthreads();
  for (int i = threadIdx.x; i < 2 * C + 1; i += blockDim.x)
    if (sd[i] != 0.0f) atomicAdd(&stats[i], sd[i]);
}

__global__ void k_bn_apply(const float* __restrict__ xin, const unsigned char* __restrict__ mask,
                           const float* __restrict__ stats, const float* __restrict__ gamma,
                           const float* __restrict__ beta, int Nvox, int C,
                           _Float16* __restrict__ outH) {
  int tid = blockIdx.x * blockDim.x + threadIdx.x;
  if (tid >= Nvox * C) return;
  int v = tid / C;
  int c = tid % C;
  float n = stats[2 * C];
  n = n > 1.0f ? n : 1.0f;
  float mean = stats[c] / n;
  float var = stats[C + c] / n - mean * mean;
  var = var > 0.0f ? var : 0.0f;
  float h = (xin[tid] - mean) * rsqrtf(var + 1e-4f) * gamma[c] + beta[c];
  h = h > 0.0f ? h : 0.0f;
  h *= mask[v] ? 1.0f : 0.0f;
  outH[tid] = (_Float16)h;
}

// =====================================================================
// weight repack: f32 [27][Cin][Cout] -> f16 B-fragment blocks.
// Block (t, kc, nt) = 32 lanes x 16 halfs, lane-contiguous so the conv
// kernel does a single 32B load per lane.
//   lane < 16 : N = lane,      K = half<8 ? half : 8 + half
//   lane >= 16: N = lane - 16, K = half<8 ? 8 + half : 16 + half
// =====================================================================
__global__ void k_pack_w(const float* __restrict__ w, int Cin, int Cout,
                         _Float16* __restrict__ wpk) {
  const int KC = Cin >> 5, NT = Cout >> 4;
  int total = 27 * KC * NT * 512;
  int tid = blockIdx.x * blockDim.x + threadIdx.x;
  if (tid >= total) return;
  int within = tid & 511;
  int blk = tid >> 9;
  int lane = within >> 4, h = within & 15;
  int nt = blk % NT;
  int b2 = blk / NT;
  int kc = b2 % KC;
  int t = b2 / KC;
  int ci = kc * 32 + ((h & 8) << 1) + ((lane >> 4) << 3) + (h & 7);
  int co = nt * 16 + (lane & 15);
  wpk[tid] = (_Float16)w[(t * Cin + ci) * Cout + co];
}

// =====================================================================
// implicit-GEMM conv via v_wmma_f32_16x16x32_f16, LDS-staged A.
// Block = 64 threads (2 waves) owning ONE 16-voxel mTile.
// Per tap: block stages the 16 input rows (Cin halfs each) into LDS once
// (global_load_async_to_lds_b128 when available), then both waves sweep
// N-tiles (nt = wave + 2*i, i < NTI), A-fragments read from LDS
// (ds_load_b128), B-fragments are pre-packed lane-contiguous 32B loads.
// Post-BN-ReLU f16 inputs are zero at inactive sites => subm semantics;
// outputs are masked; accum=1 adds (residual fusion).
// =====================================================================
template <int KC, int NTI>
__global__ __launch_bounds__(64)
void k_conv_wmma(const _Float16* __restrict__ in, int Din,
                 const _Float16* __restrict__ wpk,
                 const unsigned char* __restrict__ omask,
                 int Dout, float* __restrict__ out,
                 int stride, int pad, int accum) {
  constexpr int Cin = KC * 32;
  constexpr int NT = NTI * 2;
  constexpr int Cout = NT * 16;
  __shared__ _Float16 sA[16 * Cin];

  const int D3o = Dout * Dout * Dout;
  const int Nvox = 2 * D3o;
  const int inD3 = Din * Din * Din;
  const int mt = blockIdx.x;
  const int tid = threadIdx.x;
  const int lane = tid & 31;
  const int wave = tid >> 5;

  // ---- coordinates of the rows this thread stages (KC 16B-chunks each) ----
  int rRow[KC], rC8[KC], rB[KC], rZ[KC], rY[KC], rX[KC];
  bool rValid[KC];
#pragma unroll
  for (int j = 0; j < KC; ++j) {
    int chunkId = tid + 64 * j;        // 64*KC chunks = 16 rows * 4*KC
    int row = chunkId / (4 * KC);
    rC8[j] = chunkId - row * (4 * KC);
    rRow[j] = row;
    int v = mt * 16 + row;
    rValid[j] = v < Nvox;
    int vv = rValid[j] ? v : 0;
    int b = vv / D3o;
    int r = vv - b * D3o;
    int z = r / (Dout * Dout); r -= z * Dout * Dout;
    rB[j] = b; rZ[j] = z; rY[j] = r / Dout; rX[j] = r - (r / Dout) * Dout;
  }

  v8f acc[NTI] = {};

  for (int t = 0; t < 27; ++t) {
    const int dz = t / 9, dy = (t / 3) % 3, dx = t % 3;
    __syncthreads();  // LDS reads from previous tap are done
    // ---- stage A rows for this tap ----
#pragma unroll
    for (int j = 0; j < KC; ++j) {
      int iz = rZ[j] * stride + dz - pad;
      int iy = rY[j] * stride + dy - pad;
      int ix = rX[j] * stride + dx - pad;
      bool inb = rValid[j] && (unsigned)iz < (unsigned)Din &&
                 (unsigned)iy < (unsigned)Din && (unsigned)ix < (unsigned)Din;
      _Float16* sp = &sA[rRow[j] * Cin + rC8[j] * 8];
      if (inb) {
        const _Float16* gp =
            in + ((rB[j] * inD3 + (iz * Din + iy) * Din + ix) * Cin + rC8[j] * 8);
        async_copy16(gp, sp);
      } else {
        h8 zz = {};
        *(h8*)sp = zz;
      }
    }
    async_wait_all();
    __syncthreads();

    // prefetch next tap's B fragments
    __builtin_prefetch(wpk + ((t + 1) * KC * NT) * 512 + lane * 16, 0, 0);

    // ---- compute ----
    const _Float16* arow = &sA[(lane & 15) * Cin + (lane >> 4) * 8];
#pragma unroll
    for (int kc = 0; kc < KC; ++kc) {
      h8 lo = *(const h8*)(arow + kc * 32);
      h8 hi = *(const h8*)(arow + kc * 32 + 16);
      v16h afrag = __builtin_shufflevector(lo, hi, 0, 1, 2, 3, 4, 5, 6, 7,
                                           8, 9, 10, 11, 12, 13, 14, 15);
#pragma unroll
      for (int i = 0; i < NTI; ++i) {
        const int nt = wave + 2 * i;
        const _Float16* bp = wpk + ((t * KC + kc) * NT + nt) * 512 + lane * 16;
        v16h bfrag = *(const v16h*)bp;
        acc[i] = __builtin_amdgcn_wmma_f32_16x16x32_f16(
            false, afrag, false, bfrag, (short)0, acc[i], false, false);
      }
    }
  }

  // C/D layout: VGPR r, lanes 0-15: (M=r, N=lane); lanes 16-31: (M=8+r, N=lane-16)
  const int ncol = lane & 15;
  const int mbase = (lane >> 4) * 8;
#pragma unroll
  for (int i = 0; i < NTI; ++i) {
    const int nt = wave + 2 * i;
#pragma unroll
    for (int r = 0; r < 8; ++r) {
      int vv = mt * 16 + mbase + r;
      if (vv < Nvox) {
        float mval = omask[vv] ? 1.0f : 0.0f;
        int oi = vv * Cout + nt * 16 + ncol;
        float val = acc[i][r] * mval;
        if (accum) out[oi] += val; else out[oi] = val;
      }
    }
  }
}

// =====================================================================
// head: bn_final output h5 (f16, level5) -> x[1:], label embed, MLP
// =====================================================================
__global__ __launch_bounds__(256)
void k_head(const _Float16* __restrict__ h5, const int* __restrict__ labels,
            const float* __restrict__ le_w, const float* __restrict__ le_b,
            const float* __restrict__ p1_w, const float* __restrict__ p1_b,
            const float* __restrict__ p2_w, const float* __restrict__ p2_b,
            float* __restrict__ out) {
  __shared__ float cat[224];
  __shared__ float hv[64];
  int t = threadIdx.x;
  if (t < 160) cat[t] = (float)h5[160 + t];  // batch element 1
  if (t >= 160 && t < 224) {
    int lab = labels[0];
    int j = t - 160;
    float v = le_w[lab * 64 + j] + le_b[j];
    cat[t] = v > 0.0f ? v : 0.0f;
  }
  __syncthreads();
  if (t < 64) {
    float s = p1_b[t];
    for (int i = 0; i < 224; ++i) s = fmaf(cat[i], p1_w[i * 64 + t], s);
    hv[t] = s > 0.0f ? s : 0.0f;
  }
  __syncthreads();
  if (t == 0) {
    float s = p2_b[0];
    for (int j = 0; j < 64; ++j) s = fmaf(hv[j], p2_w[j], s);
    out[0] = s;
  }
}

// =====================================================================
// host orchestration
// =====================================================================
static inline int idiv(long a, long b) { return (int)((a + b - 1) / b); }

extern "C" void kernel_launch(void* const* d_in, const int* in_sizes, int n_in,
                              void* d_out, int out_size, void* d_ws, size_t ws_size,
                              hipStream_t stream) {
  (void)in_sizes; (void)n_in; (void)out_size; (void)ws_size;
  const int*   coords = (const int*)d_in[0];
  const float* colors = (const float*)d_in[1];
  const int*   labels = (const int*)d_in[2];
  // params flattened as a JAX pytree (dict keys sorted):
  // bn_final(g,b), conv0, le_b, le_w, p1_b, p1_w, p2_b, p2_w, stages[0..4]
  const float* bnf_g = (const float*)d_in[3];
  const float* bnf_b = (const float*)d_in[4];
  const float* w0    = (const float*)d_in[5];
  const float* le_b  = (const float*)d_in[6];
  const float* le_w  = (const float*)d_in[7];
  const float* p1_b  = (const float*)d_in[8];
  const float* p1_w  = (const float*)d_in[9];
  const float* p2_b  = (const float*)d_in[10];
  const float* p2_w  = (const float*)d_in[11];
  // per-stage dict keys sorted: bn0,bn1,bna,bnc,ca,cb,cc,cd,short (bn = (gamma,beta))
  struct Stage { const float *bn0g,*bn0b,*bn1g,*bn1b,*bnag,*bnab,*bncg,*bncb,*ca,*cb,*cc,*cd,*shw; };
  Stage st[5];
  for (int s = 0; s < 5; ++s) {
    int base = 12 + 13 * s;
    st[s].bn0g = (const float*)d_in[base + 0];  st[s].bn0b = (const float*)d_in[base + 1];
    st[s].bn1g = (const float*)d_in[base + 2];  st[s].bn1b = (const float*)d_in[base + 3];
    st[s].bnag = (const float*)d_in[base + 4];  st[s].bnab = (const float*)d_in[base + 5];
    st[s].bncg = (const float*)d_in[base + 6];  st[s].bncb = (const float*)d_in[base + 7];
    st[s].ca = (const float*)d_in[base + 8];    st[s].cb = (const float*)d_in[base + 9];
    st[s].cc = (const float*)d_in[base + 10];   st[s].cd = (const float*)d_in[base + 11];
    st[s].shw = (const float*)d_in[base + 12];
  }

  // ---- workspace carve ----
  char* wsb = (char*)d_ws;
  size_t off = 0;
  auto alloc = [&](size_t bytes) -> void* {
    void* p = wsb + off;
    off += (bytes + 255) & ~(size_t)255;
    return p;
  };
  float* gridF = (float*)alloc((size_t)2 * S3 * 3 * 4);
  float* cnt   = (float*)alloc((size_t)2 * S3 * 4);
  unsigned char* m127 = (unsigned char*)alloc((size_t)2 * S3);

  const int D[6] = {63, 31, 15, 7, 3, 1};
  const int C[6] = {32, 32, 64, 96, 128, 160};
  float* xF[6]; _Float16* hF[6]; unsigned char* mk[6]; int NV[6];
  for (int l = 0; l < 6; ++l) {
    int d3 = D[l] * D[l] * D[l];
    NV[l] = 2 * d3;
    xF[l] = (float*)alloc((size_t)NV[l] * C[l] * 4);
    hF[l] = (_Float16*)alloc((size_t)NV[l] * C[l] * 2);
    mk[l] = (unsigned char*)alloc((size_t)NV[l]);
  }
  float* tmpF = (float*)alloc((size_t)NV[1] * C[1] * 4);
  _Float16* wpk = (_Float16*)alloc((size_t)27 * 128 * 160 * 2);
  float* stats = (float*)alloc(352 * 4);

  auto bnorm = [&](const float* xin, const unsigned char* m, int Nvox, int Cc,
                   const float* g, const float* b, _Float16* outH) {
    k_zero<<<2, 256, 0, stream>>>(stats, 352);
    k_bn_stats<<<idiv(Nvox, 64), 256, 0, stream>>>(xin, m, Nvox, Cc, stats);
    k_bn_apply<<<idiv((long)Nvox * Cc, 256), 256, 0, stream>>>(xin, m, stats, g, b, Nvox, Cc, outH);
  };
  auto conv = [&](const _Float16* in, int Din, int Cin, const float* wraw,
                  const unsigned char* om, int Dout, int Cout, float* outp,
                  int stride, int pad, int accum) {
    int KC = Cin / 32, NT = Cout / 16, NTI = NT / 2;
    long wtot = (long)27 * KC * NT * 512;
    k_pack_w<<<idiv(wtot, 256), 256, 0, stream>>>(wraw, Cin, Cout, wpk);
    int Nvox = 2 * Dout * Dout * Dout;
    int Mt = idiv(Nvox, 16);
    if (KC == 1 && NTI == 1)      k_conv_wmma<1, 1><<<Mt, 64, 0, stream>>>(in, Din, wpk, om, Dout, outp, stride, pad, accum);
    else if (KC == 1 && NTI == 2) k_conv_wmma<1, 2><<<Mt, 64, 0, stream>>>(in, Din, wpk, om, Dout, outp, stride, pad, accum);
    else if (KC == 2 && NTI == 2) k_conv_wmma<2, 2><<<Mt, 64, 0, stream>>>(in, Din, wpk, om, Dout, outp, stride, pad, accum);
    else if (KC == 2 && NTI == 3) k_conv_wmma<2, 3><<<Mt, 64, 0, stream>>>(in, Din, wpk, om, Dout, outp, stride, pad, accum);
    else if (KC == 3 && NTI == 3) k_conv_wmma<3, 3><<<Mt, 64, 0, stream>>>(in, Din, wpk, om, Dout, outp, stride, pad, accum);
    else if (KC == 3 && NTI == 4) k_conv_wmma<3, 4><<<Mt, 64, 0, stream>>>(in, Din, wpk, om, Dout, outp, stride, pad, accum);
    else if (KC == 4 && NTI == 4) k_conv_wmma<4, 4><<<Mt, 64, 0, stream>>>(in, Din, wpk, om, Dout, outp, stride, pad, accum);
    else if (KC == 4 && NTI == 5) k_conv_wmma<4, 5><<<Mt, 64, 0, stream>>>(in, Din, wpk, om, Dout, outp, stride, pad, accum);
    else                          k_conv_wmma<5, 5><<<Mt, 64, 0, stream>>>(in, Din, wpk, om, Dout, outp, stride, pad, accum);
  };

  // ---- voxelize + conv0 + pool ----
  k_zero<<<idiv(2 * S3 * 3, 256), 256, 0, stream>>>(gridF, 2 * S3 * 3);
  k_zero<<<idiv(2 * S3, 256), 256, 0, stream>>>(cnt, 2 * S3);
  k_scatter<<<idiv(NPTS, 256), 256, 0, stream>>>(coords, colors, gridF, cnt, NPTS);
  k_finalize<<<idiv(2 * S3, 256), 256, 0, stream>>>(gridF, cnt, m127, 2 * S3);
  k_conv0_pool<<<idiv((long)NV[0] * 32, 256), 256, 0, stream>>>(gridF, m127, w0, xF[0], mk[0]);

  // ---- 5 residual stages ----
  for (int s = 0; s < 5; ++s) {
    int li = s, lo = s + 1;
    bnorm(xF[li], mk[li], NV[li], C[li], st[s].bn0g, st[s].bn0b, hF[li]);
    k_pool_mask<<<idiv(NV[lo], 256), 256, 0, stream>>>(mk[li], D[li], mk[lo], D[lo]);
    conv(hF[li], D[li], C[li], st[s].shw, mk[lo], D[lo], C[lo], xF[lo], 2, 0, 0);  // short
    conv(hF[li], D[li], C[li], st[s].ca,  mk[lo], D[lo], C[lo], tmpF,   2, 0, 0);  // main
    bnorm(tmpF, mk[lo], NV[lo], C[lo], st[s].bnag, st[s].bnab, hF[lo]);
    conv(hF[lo], D[lo], C[lo], st[s].cb, mk[lo], D[lo], C[lo], xF[lo], 1, 1, 1);   // x = short+main
    bnorm(xF[lo], mk[lo], NV[lo], C[lo], st[s].bn1g, st[s].bn1b, hF[lo]);
    conv(hF[lo], D[lo], C[lo], st[s].cc, mk[lo], D[lo], C[lo], tmpF, 1, 1, 0);
    bnorm(tmpF, mk[lo], NV[lo], C[lo], st[s].bncg, st[s].bncb, hF[lo]);
    conv(hF[lo], D[lo], C[lo], st[s].cd, mk[lo], D[lo], C[lo], xF[lo], 1, 1, 1);   // x = x + h
  }

  // ---- final BN + head ----
  bnorm(xF[5], mk[5], NV[5], 160, bnf_g, bnf_b, hF[5]);
  k_head<<<1, 256, 0, stream>>>(hF[5], labels, le_w, le_b, p1_w, p1_b, p2_w, p2_b,
                                (float*)d_out);
}
